// MambaRec_30399778521711
// MI455X (gfx1250) — matrix operand order repaired and checked by reading, not compiled, not run on hardware
//
#include <hip/hip_runtime.h>
#include <hip/hip_bf16.h>
#include <math.h>

// ---- problem constants (match reference) ----
#define HH       64
#define DSTATE   32
#define DCONV    4
#define DIN      128
#define DTRANK   4
#define LSEQ     200
#define LPAD     208      // 13 * 16
#define NBATCH   1024

typedef __bf16 bf16;
typedef __attribute__((ext_vector_type(16))) __bf16 v16bf;
typedef __attribute__((ext_vector_type(8)))  __bf16 v8bf;
typedef __attribute__((ext_vector_type(8)))  float  v8f;

// LDS layout (bytes); total 163072 < 160 KB -> 2 workgroups per WGP (4 waves/SIMD)
//   [OFF_U]   bf16 [LPAD][DIN] 53248 : seq bf16 [LPAD][HH] (ph1-2) -> u (ph3+) -> y in place
//   [OFF_Z]   bf16 [LPAD][DIN] 53248 : z
//   [OFF_X]   bf16 [LPAD][DIN] 53248 : xp (ph2-3) -> Bm/Cm f32 (ph4+) -> lf f32 (ph7+)
//   [OFF_DTR] f32  [LPAD][4]    3328 : dt_raw (ph4+), consumed on the fly in the scan
#define OFF_U    0
#define OFF_Z    53248
#define OFF_X    106496
#define OFF_DTR  159744
#define SMEM_BYTES 163072

// ---- WMMA helpers ----
__device__ inline v8f wmma_bf16(v16bf a, v16bf b, v8f c) {
  // (neg_a, A, neg_b, B, c_mod, C, reuse_a, reuse_b)
  return __builtin_amdgcn_wmma_f32_16x16x32_bf16(false, a, false, b, (short)0, c,
                                                 false, false);
}

// A fragment: 16x32 bf16 from LDS, row-major with leading dim `ld`.
// ISA layout: lanes 0-15 -> M=0..15, K = {0..7, 16..23}; lanes 16-31 -> K = {8..15, 24..31}
__device__ inline v16bf load_a_lds(const bf16* base, int ld, int m0, int k0) {
  const int lane = threadIdx.x & 31;
  const int m  = m0 + (lane & 15);
  const int kb = k0 + ((lane >> 4) << 3);
  union { v16bf v; v8bf h[2]; } f;
  f.h[0] = *(const v8bf*)(base + m * ld + kb);
  f.h[1] = *(const v8bf*)(base + m * ld + kb + 16);
  return f.v;
}

// B fragment: 32x16 (K x N) built from global f32 weight, W[n][k] row-major, ld = K stride.
// ISA layout: lane n = lane&15, K = {0..7, 16..23} (lanes 0-15), {8..15, 24..31} (lanes 16-31).
__device__ inline v16bf load_b_gl(const float* __restrict__ W, int ld, int n0, int k0,
                                  int nmax) {
  const int lane = threadIdx.x & 31;
  const int n  = n0 + (lane & 15);
  const int kb = k0 + ((lane >> 4) << 3);
  union { v16bf v; bf16 e[16]; } f;
  if (n < nmax) {
    const float* p = W + n * ld + kb;
#pragma unroll
    for (int j = 0; j < 16; ++j) f.e[j] = (bf16)p[(j & 7) + ((j >> 3) << 4)];
  } else {
#pragma unroll
    for (int j = 0; j < 16; ++j) f.e[j] = (bf16)0.0f;
  }
  return f.v;
}

__global__ void __launch_bounds__(256, 1)
mamba_fused(const int* __restrict__ log_seqs, const int* __restrict__ pos_seqs,
            const int* __restrict__ neg_seqs,
            const float* __restrict__ item_emb, const float* __restrict__ pos_emb,
            const float* __restrict__ W_in, const float* __restrict__ conv_w,
            const float* __restrict__ conv_b, const float* __restrict__ W_xproj,
            const float* __restrict__ W_dt, const float* __restrict__ b_dt,
            const float* __restrict__ A_log, const float* __restrict__ D_skip,
            const float* __restrict__ W_out, float* __restrict__ out) {
  extern __shared__ char smem[];
  bf16*  s_u   = (bf16*)(smem + OFF_U);               // also seq in ph1-2, y after scan
  bf16*  s_seq = (bf16*)(smem + OFF_U);               // [LPAD][HH]
  bf16*  s_z   = (bf16*)(smem + OFF_Z);
  bf16*  s_xp  = (bf16*)(smem + OFF_X);
  float* s_Bm  = (float*)(smem + OFF_X);              // [LPAD][32] after conv
  float* s_Cm  = (float*)(smem + OFF_X + 26624);      // [LPAD][32]
  float* s_lf  = (float*)(smem + OFF_X);              // [LPAD][HH] after scan
  float* s_dtr = (float*)(smem + OFF_DTR);            // [LPAD][4]

  const int b    = blockIdx.x;
  const int tid  = threadIdx.x;
  const int wave = tid >> 5;
  const int lane = tid & 31;

  // -------- Phase 1: gather embeddings -> s_seq (bf16), zero-padded rows --------
  for (int i = tid; i < LPAD * HH; i += 256) {
    const int l = i >> 6, h = i & 63;
    float v = 0.0f;
    if (l < LSEQ) {
      const int it = log_seqs[b * LSEQ + l];
      v = item_emb[it * HH + h] * 8.0f + pos_emb[l * HH + h];
    }
    s_seq[i] = (bf16)v;
  }
  __syncthreads();

  // -------- Phase 2: xz = seq @ W_in^T  (M=208, K=64, N=256) via WMMA --------
  for (int nt = wave; nt < 16; nt += 8) {
    const v16bf b0 = load_b_gl(W_in, HH, nt * 16, 0, 256);
    const v16bf b1 = load_b_gl(W_in, HH, nt * 16, 32, 256);
    for (int mt = 0; mt < 13; ++mt) {
      const v16bf a0 = load_a_lds(s_seq, HH, mt * 16, 0);
      const v16bf a1 = load_a_lds(s_seq, HH, mt * 16, 32);
      v8f c = {};
      c = wmma_bf16(a0, b0, c);
      c = wmma_bf16(a1, b1, c);
      const int col = nt * 16 + (lane & 15);
      const int rb  = mt * 16 + ((lane >> 4) << 3);
      bf16* dst = (col < DIN) ? (s_xp + col) : (s_z + (col - DIN));
#pragma unroll
      for (int r = 0; r < 8; ++r) dst[(rb + r) * DIN] = (bf16)c[r];
    }
  }
  __syncthreads();

  // -------- Phase 3: causal depthwise conv(4) + SiLU -> s_u (zero pad rows) ----
  // (s_u overwrites the dead seq region; barrier above guarantees GEMM1 is done)
  {
    const int d  = tid & 127;
    const float cb = conv_b[d];
    float cw[4];
#pragma unroll
    for (int j = 0; j < 4; ++j) cw[j] = conv_w[d * 4 + j];
    for (int i = tid; i < LPAD * DIN; i += 256) {
      const int l = i >> 7;
      float uv = 0.0f;
      if (l < LSEQ) {
        float acc = cb;
#pragma unroll
        for (int j = 0; j < 4; ++j) {
          const int li = l - 3 + j;
          if (li >= 0) acc += cw[j] * (float)s_xp[li * DIN + d];
        }
        uv = acc / (1.0f + __expf(-acc));
      }
      s_u[i] = (bf16)uv;
    }
  }
  __syncthreads();

  // -------- Phase 4: x_dbl = u @ W_xproj^T (M=208, K=128, N=68->80) via WMMA ---
  // route columns: [0,4) dt_raw, [4,36) Bm, [36,68) Cm   (overwrites s_xp)
  for (int t = wave; t < 13 * 5; t += 8) {
    const int mt = t / 5, nt = t % 5;
    v8f c = {};
#pragma unroll
    for (int kt = 0; kt < 4; ++kt) {
      const v16bf a  = load_a_lds(s_u, DIN, mt * 16, kt * 32);
      const v16bf bb = load_b_gl(W_xproj, DIN, nt * 16, kt * 32, DTRANK + 2 * DSTATE);
      c = wmma_bf16(a, bb, c);
    }
    const int col = nt * 16 + (lane & 15);
    const int rb  = mt * 16 + ((lane >> 4) << 3);
#pragma unroll
    for (int r = 0; r < 8; ++r) {
      const int row = rb + r;
      const float v = c[r];
      if (col < DTRANK)                    s_dtr[row * DTRANK + col] = v;
      else if (col < DTRANK + DSTATE)      s_Bm[row * DSTATE + (col - DTRANK)] = v;
      else if (col < DTRANK + 2 * DSTATE)  s_Cm[row * DSTATE + (col - DTRANK - DSTATE)] = v;
    }
  }
  __syncthreads();

  // -------- Phase 5+6: selective scan over L, dt fused in (barrier-free) -------
  // thread t owns d = t>>1, half the 32 states (nh = t&1); partner is lane^1
  // (same wave), so the cross-thread y reduction is a single shfl_xor and the
  // whole 200-step scan needs no block barriers. dt = softplus(dtr.Wdt + b_dt)
  // is computed on the fly in f32 (no bf16 round trip).
  {
    const int d  = tid >> 1;
    const int nh = tid & 1;
    float Ar[16];
#pragma unroll
    for (int i = 0; i < 16; ++i)
      Ar[i] = -__expf(A_log[d * DSTATE + nh * 16 + i]);
    float wdt[4];
#pragma unroll
    for (int r = 0; r < 4; ++r) wdt[r] = W_dt[d * 4 + r];
    const float bd  = b_dt[d];
    const float dsk = D_skip[d];
    float hs[16];
#pragma unroll
    for (int i = 0; i < 16; ++i) hs[i] = 0.0f;
    for (int l = 0; l < LSEQ; ++l) {
      float x = bd;
#pragma unroll
      for (int r = 0; r < 4; ++r) x += wdt[r] * s_dtr[l * DTRANK + r];
      const float dtv = (x > 20.0f) ? x : log1pf(__expf(x));
      const float uv  = (float)s_u[l * DIN + d];
      const float zv  = (float)s_z[l * DIN + d];
      float acc = 0.0f;
#pragma unroll
      for (int i = 0; i < 16; ++i) {
        const float e = __expf(dtv * Ar[i]);
        hs[i] = e * hs[i] + dtv * s_Bm[l * DSTATE + nh * 16 + i] * uv;
        acc += hs[i] * s_Cm[l * DSTATE + nh * 16 + i];
      }
      acc += __shfl_xor(acc, 1, 32);
      if (nh == 0) {
        float y = acc + uv * dsk;
        y *= zv / (1.0f + __expf(-zv));     // y * silu(z)
        s_u[l * DIN + d] = (bf16)y;          // overwrite u row l in place
      }
    }
  }
  __syncthreads();

  // -------- Phase 7: log_feats = y @ W_out^T (M=208, K=128, N=64) via WMMA -----
  for (int t = wave; t < 13 * 4; t += 8) {
    const int mt = t >> 2, nt = t & 3;
    v8f c = {};
#pragma unroll
    for (int kt = 0; kt < 4; ++kt) {
      const v16bf a  = load_a_lds(s_u, DIN, mt * 16, kt * 32);
      const v16bf bb = load_b_gl(W_out, DIN, nt * 16, kt * 32, HH);
      c = wmma_bf16(a, bb, c);
    }
    const int col = nt * 16 + (lane & 15);
    const int rb  = mt * 16 + ((lane >> 4) << 3);
#pragma unroll
    for (int r = 0; r < 8; ++r) s_lf[(rb + r) * HH + col] = c[r];
  }
  __syncthreads();

  // -------- Phase 8: pos/neg logits -------------------------------------------
  for (int p = tid; p < LSEQ * 2; p += 256) {
    const int l = p >> 1, which = p & 1;
    const int idx = which ? neg_seqs[b * LSEQ + l] : pos_seqs[b * LSEQ + l];
    const float* ev = item_emb + (long long)idx * HH;
    const float* fv = s_lf + l * HH;
    float acc = 0.0f;
#pragma unroll 16
    for (int h = 0; h < HH; ++h) acc += fv[h] * ev[h];
    out[(long long)which * (NBATCH * LSEQ) + b * LSEQ + l] = acc;
  }
}

extern "C" void kernel_launch(void* const* d_in, const int* in_sizes, int n_in,
                              void* d_out, int out_size, void* d_ws, size_t ws_size,
                              hipStream_t stream) {
  // setup_inputs order:
  // 0 user_ids (unused), 1 log_seqs, 2 pos_seqs, 3 neg_seqs, 4 item_emb, 5 pos_emb,
  // 6 W_in, 7 conv_w, 8 conv_b, 9 W_xproj, 10 W_dt, 11 b_dt, 12 A_log, 13 D_skip, 14 W_out
  const int*   log_seqs = (const int*)d_in[1];
  const int*   pos_seqs = (const int*)d_in[2];
  const int*   neg_seqs = (const int*)d_in[3];
  const float* item_emb = (const float*)d_in[4];
  const float* pos_emb  = (const float*)d_in[5];
  const float* W_in     = (const float*)d_in[6];
  const float* conv_w   = (const float*)d_in[7];
  const float* conv_b   = (const float*)d_in[8];
  const float* W_xproj  = (const float*)d_in[9];
  const float* W_dt     = (const float*)d_in[10];
  const float* b_dt     = (const float*)d_in[11];
  const float* A_log    = (const float*)d_in[12];
  const float* D_skip   = (const float*)d_in[13];
  const float* W_out    = (const float*)d_in[14];
  float* out = (float*)d_out;

  (void)in_sizes; (void)n_in; (void)out_size; (void)d_ws; (void)ws_size;

  hipFuncSetAttribute((const void*)mamba_fused,
                      hipFuncAttributeMaxDynamicSharedMemorySize, SMEM_BYTES);
  mamba_fused<<<NBATCH, 256, SMEM_BYTES, stream>>>(
      log_seqs, pos_seqs, neg_seqs, item_emb, pos_emb, W_in, conv_w, conv_b,
      W_xproj, W_dt, b_dt, A_log, D_skip, W_out, out);
}
